// MultiHeadedAttentionWithRope_6631429505312
// MI455X (gfx1250) — matrix operand order repaired and checked by reading, not compiled, not run on hardware
//
#include <hip/hip_runtime.h>
#include <stdint.h>

// MultiHeadedAttentionWithRope for MI455X (gfx1250, wave32, WMMA bf16).
//   x:(4,2048,1024) f32, q_w/k_w/v_w/o_w:(1024,1024) f32 -> out (4,2048,1024) f32
// Pipeline: 3x WMMA GEMM (x@W^T, bf16 in/f32 acc) -> RoPE(Q,K) ->
//           flash attention (bf16 WMMA scores + P@V, online softmax) ->
//           WMMA GEMM with o_w -> f32 out.

#define BATCH  4
#define SEQ    2048
#define DMODEL 1024
#define NHEAD  16
#define DHEAD  64
#define MROWS  (BATCH * SEQ)   // 8192

typedef __attribute__((ext_vector_type(16))) __bf16 bf16x16;
typedef __attribute__((ext_vector_type(8)))  float  f32x8;

struct FragU { union { bf16x16 v; uint32_t u[8]; }; };

__device__ __forceinline__ uint16_t f2bf(float f) {
  union { float f; uint32_t u; } c; c.f = f;
  uint32_t u = c.u;
  return (uint16_t)((u + 0x7FFFu + ((u >> 16) & 1u)) >> 16);  // RNE
}
__device__ __forceinline__ float bf2f(uint16_t h) {
  union { float f; uint32_t u; } c; c.u = ((uint32_t)h) << 16; return c.f;
}
__device__ __forceinline__ uint32_t pack2bf(float a, float b) {
  return (uint32_t)f2bf(a) | ((uint32_t)f2bf(b) << 16);
}
__device__ __forceinline__ f32x8 wmma_bf16(const FragU& a, const FragU& b, f32x8 c) {
  return __builtin_amdgcn_wmma_f32_16x16x32_bf16(false, a.v, false, b.v,
                                                 (short)0, c, false, false);
}
// Load two aligned 16B chunks into one 16-element bf16 fragment.
__device__ __forceinline__ void ld2x16(FragU& f, const void* p0, const void* p1) {
  uint4 a = *(const uint4*)p0;
  uint4 b = *(const uint4*)p1;
  f.u[0]=a.x; f.u[1]=a.y; f.u[2]=a.z; f.u[3]=a.w;
  f.u[4]=b.x; f.u[5]=b.y; f.u[6]=b.z; f.u[7]=b.w;
}

// ---------------------------------------------------------------------------
// GEMM: out[gm][gn] = sum_k A[gm][k] * W[gn][k]   (A: MROWSxDMODEL, W row-major)
// Block: 128 thr (4 waves), tile 64x64, K-step 32. LDS-staged bf16.
// mode 0: bf16 -> [b,h,s,dh]   (Q or K)
// mode 1: bf16 -> [b,h,dh,s]   (V transposed for attention B-fragments)
// mode 2: f32  -> [gm][DMODEL] (final output)
// ---------------------------------------------------------------------------
__global__ __launch_bounds__(128) void gemm_bf16_wmma(
    const void* __restrict__ Aptr, int a_is_bf16,
    const float* __restrict__ W,
    void* __restrict__ Dst, int mode)
{
  __shared__ alignas(16) uint32_t As[64 * 16];  // 64 rows x 32 bf16 (as u32 pairs)
  __shared__ alignas(16) uint32_t Bs[64 * 16];
  const int tid  = threadIdx.x;
  const int lane = tid & 31;
  const int wv   = tid >> 5;
  const int ln15 = lane & 15;
  const int hi   = lane >> 4;
  const int rowbase = blockIdx.y * 64;
  const int colbase = blockIdx.x * 64;
  const int r  = tid >> 1;     // 0..63 staged row
  const int hf = tid & 1;      // k half (16 elems)

  f32x8 acc[4];
  #pragma unroll
  for (int i = 0; i < 4; ++i)
    #pragma unroll
    for (int j = 0; j < 8; ++j) acc[i][j] = 0.0f;

  for (int k0 = 0; k0 < DMODEL; k0 += 32) {
    // ---- stage A tile (64x32) as bf16 ----
    if (a_is_bf16) {
      const uint16_t* src = (const uint16_t*)Aptr + (size_t)(rowbase + r) * DMODEL + k0 + hf * 16;
      uint4 p0 = *(const uint4*)src;
      uint4 p1 = *(const uint4*)(src + 8);
      uint32_t* d = &As[r * 16 + hf * 8];
      d[0]=p0.x; d[1]=p0.y; d[2]=p0.z; d[3]=p0.w;
      d[4]=p1.x; d[5]=p1.y; d[6]=p1.z; d[7]=p1.w;
      if (k0 + 32 < DMODEL) __builtin_prefetch(src + 32, 0, 1);
    } else {
      const float* src = (const float*)Aptr + (size_t)(rowbase + r) * DMODEL + k0 + hf * 16;
      const float4* s4 = (const float4*)src;
      float4 f0 = s4[0], f1 = s4[1], f2 = s4[2], f3 = s4[3];
      uint32_t* d = &As[r * 16 + hf * 8];
      d[0]=pack2bf(f0.x,f0.y); d[1]=pack2bf(f0.z,f0.w);
      d[2]=pack2bf(f1.x,f1.y); d[3]=pack2bf(f1.z,f1.w);
      d[4]=pack2bf(f2.x,f2.y); d[5]=pack2bf(f2.z,f2.w);
      d[6]=pack2bf(f3.x,f3.y); d[7]=pack2bf(f3.z,f3.w);
      if (k0 + 32 < DMODEL) __builtin_prefetch(src + 32, 0, 1);
    }
    // ---- stage B tile (64 output cols x 32) from weights ----
    {
      const float* src = W + (size_t)(colbase + r) * DMODEL + k0 + hf * 16;
      const float4* s4 = (const float4*)src;
      float4 f0 = s4[0], f1 = s4[1], f2 = s4[2], f3 = s4[3];
      uint32_t* d = &Bs[r * 16 + hf * 8];
      d[0]=pack2bf(f0.x,f0.y); d[1]=pack2bf(f0.z,f0.w);
      d[2]=pack2bf(f1.x,f1.y); d[3]=pack2bf(f1.z,f1.w);
      d[4]=pack2bf(f2.x,f2.y); d[5]=pack2bf(f2.z,f2.w);
      d[6]=pack2bf(f3.x,f3.y); d[7]=pack2bf(f3.z,f3.w);
      if (k0 + 32 < DMODEL) __builtin_prefetch(src + 32, 0, 1);
    }
    __syncthreads();

    // A fragment: lanes 0-15 hold K=0-7,16-23; lanes 16-31 hold K=8-15,24-31
    FragU a;
    {
      const uint32_t* ab = &As[(wv * 16 + ln15) * 16 + hi * 4];
      ld2x16(a, ab, ab + 8);
    }
    #pragma unroll
    for (int ns = 0; ns < 4; ++ns) {
      // B fragment: lane=column, lanes 0-15 K=0-7,16-23; lanes 16-31 K=8-15,24-31
      FragU b;
      const uint32_t* bb = &Bs[(ns * 16 + ln15) * 16 + hi * 4];
      ld2x16(b, bb, bb + 8);
      acc[ns] = wmma_bf16(a, b, acc[ns]);
    }
    __syncthreads();
  }

  // ---- epilogue: C/D layout: VGPR v -> M = v + 8*hi, N = ln15 ----
  #pragma unroll
  for (int ns = 0; ns < 4; ++ns) {
    #pragma unroll
    for (int v = 0; v < 8; ++v) {
      const int gm = rowbase + wv * 16 + v + hi * 8;
      const int gn = colbase + ns * 16 + ln15;
      const float val = acc[ns][v];
      if (mode == 2) {
        ((float*)Dst)[(size_t)gm * DMODEL + gn] = val;
      } else {
        const int b = gm >> 11, s = gm & (SEQ - 1);
        const int h = gn >> 6,  d = gn & (DHEAD - 1);
        const uint16_t o = f2bf(val);
        if (mode == 0)
          ((uint16_t*)Dst)[(((size_t)(b * NHEAD + h) * SEQ + s) << 6) + d] = o;
        else
          ((uint16_t*)Dst)[(((size_t)(b * NHEAD + h) * DHEAD + d) << 11) + s] = o;
      }
    }
  }
}

// ---------------------------------------------------------------------------
// RoPE in-place on a [64][SEQ][DHEAD] bf16 buffer (64 = b*h).
// ---------------------------------------------------------------------------
__global__ __launch_bounds__(256) void rope_kernel(uint16_t* __restrict__ t) {
  const int gid = blockIdx.x * blockDim.x + threadIdx.x;  // 64*SEQ*32 threads
  const int i  = gid & 31;
  const int s  = (gid >> 5) & (SEQ - 1);
  const int bh = gid >> 16;
  const size_t off = ((size_t)bh * SEQ + s) * DHEAD + 2 * i;
  const float x1 = bf2f(t[off]), x2 = bf2f(t[off + 1]);
  // freq = theta^(-i/32) = exp(-i*ln(10000)/32)
  const float freq = __expf(-(float)i * (9.210340371976184f / 32.0f));
  const float ang = (float)s * freq;
  const float c = cosf(ang), sn = sinf(ang);
  t[off]     = f2bf(x1 * c - x2 * sn);
  t[off + 1] = f2bf(x1 * sn + x2 * c);
}

// ---------------------------------------------------------------------------
// Flash attention: one wave per (b,h,16-query tile). Keys in blocks of 32.
// scores = Q(16x64) @ K^T  (2 WMMAs per 16-key subtile), online softmax,
// O += P(16x32) @ V (4 WMMAs over dh subtiles; V pre-transposed [dh][s]).
// ---------------------------------------------------------------------------
__global__ __launch_bounds__(32) void attn_kernel(
    const uint16_t* __restrict__ Q, const uint16_t* __restrict__ K,
    const uint16_t* __restrict__ Vt, uint16_t* __restrict__ Obuf)
{
  __shared__ alignas(16) uint32_t Pl[16 * 16];  // 16x32 bf16 probs (C->A relayout)
  const int lane = threadIdx.x;
  const int ln15 = lane & 15;
  const int hi   = lane >> 4;
  const int qt = blockIdx.x & 127;      // query tile (SEQ/16 = 128)
  const int bh = blockIdx.x >> 7;       // 0..63
  const int b  = bh >> 4;
  const int h  = bh & 15;
  const size_t hbase = (size_t)bh * SEQ * DHEAD;

  // Q fragments (A layout), reused across all key blocks.
  FragU aq0, aq1;
  {
    const uint16_t* qrow = Q + hbase + (size_t)(qt * 16 + ln15) * DHEAD;
    const int kh = hi * 8;
    ld2x16(aq0, qrow + kh,      qrow + kh + 16);       // d = 0..31 pattern
    ld2x16(aq1, qrow + 32 + kh, qrow + 32 + kh + 16);  // d = 32..63 pattern
  }

  f32x8 acc[4];
  #pragma unroll
  for (int i = 0; i < 4; ++i)
    #pragma unroll
    for (int j = 0; j < 8; ++j) acc[i][j] = 0.0f;
  float mrow[8], lrow[8];
  #pragma unroll
  for (int v = 0; v < 8; ++v) { mrow[v] = -1e30f; lrow[v] = 0.0f; }

  const int nkb = ((qt + 1) * 16 + 31) >> 5;  // causal: key blocks needed
  for (int kb = 0; kb < nkb; ++kb) {
    float sc[2][8];
    // ---- scores for the two 16-key subtiles ----
    #pragma unroll
    for (int sub = 0; sub < 2; ++sub) {
      const uint16_t* krow = K + hbase +
          (size_t)(kb * 32 + sub * 16 + ln15) * DHEAD + hi * 8;
      FragU bk0, bk1;                      // B layout: k = d
      ld2x16(bk0, krow,      krow + 16);   // d 0..31
      ld2x16(bk1, krow + 32, krow + 48);   // d 32..63
      f32x8 c;
      #pragma unroll
      for (int j = 0; j < 8; ++j) c[j] = 0.0f;
      c = wmma_bf16(aq0, bk0, c);
      c = wmma_bf16(aq1, bk1, c);
      const int kj = kb * 32 + sub * 16 + ln15;
      #pragma unroll
      for (int v = 0; v < 8; ++v) {
        const int qi = qt * 16 + v + hi * 8;
        const float xv = c[v] * 0.125f;             // 1/sqrt(64)
        sc[sub][v] = (kj > qi) ? -1e30f : xv;       // causal mask
      }
    }
    // ---- online softmax update (row reductions within 16-lane halves) ----
    float alpha[8], p2[2][8];
    #pragma unroll
    for (int v = 0; v < 8; ++v) {
      float mx = fmaxf(sc[0][v], sc[1][v]);
      mx = fmaxf(mx, __shfl_xor(mx, 1, 32));
      mx = fmaxf(mx, __shfl_xor(mx, 2, 32));
      mx = fmaxf(mx, __shfl_xor(mx, 4, 32));
      mx = fmaxf(mx, __shfl_xor(mx, 8, 32));
      const float mnew = fmaxf(mrow[v], mx);
      alpha[v] = __expf(mrow[v] - mnew);
      mrow[v] = mnew;
      const float p0 = __expf(sc[0][v] - mnew);
      const float p1 = __expf(sc[1][v] - mnew);
      p2[0][v] = p0; p2[1][v] = p1;
      float sum = p0 + p1;
      sum += __shfl_xor(sum, 1, 32);
      sum += __shfl_xor(sum, 2, 32);
      sum += __shfl_xor(sum, 4, 32);
      sum += __shfl_xor(sum, 8, 32);
      lrow[v] = alpha[v] * lrow[v] + sum;
    }
    #pragma unroll
    for (int f = 0; f < 4; ++f)
      #pragma unroll
      for (int v = 0; v < 8; ++v) acc[f][v] *= alpha[v];

    // ---- P: C layout -> LDS -> A layout ----
    uint16_t* Pl16 = (uint16_t*)Pl;
    #pragma unroll
    for (int sub = 0; sub < 2; ++sub)
      #pragma unroll
      for (int v = 0; v < 8; ++v)
        Pl16[(v + hi * 8) * 32 + sub * 16 + ln15] = f2bf(p2[sub][v]);
    __syncthreads();
    FragU ap;
    {
      const uint32_t* pr = &Pl[ln15 * 16 + hi * 4];
      ld2x16(ap, pr, pr + 8);
    }
    __syncthreads();

    // ---- O += P @ V  (V transposed: rows = dh, contiguous over keys) ----
    #pragma unroll
    for (int f = 0; f < 4; ++f) {
      const uint16_t* vrow = Vt + (size_t)bh * DHEAD * SEQ +
          (size_t)(f * 16 + ln15) * SEQ + kb * 32 + hi * 8;
      FragU bv;                            // B layout: k = key index
      ld2x16(bv, vrow, vrow + 16);
      acc[f] = wmma_bf16(ap, bv, acc[f]);
    }
  }

  // ---- normalize and scatter back to [b, s, h*64+d] bf16 ----
  #pragma unroll
  for (int f = 0; f < 4; ++f)
    #pragma unroll
    for (int v = 0; v < 8; ++v) {
      const int srow = qt * 16 + v + hi * 8;
      const int d = f * 16 + ln15;
      const float o = acc[f][v] / lrow[v];
      Obuf[((size_t)(b * SEQ + srow)) * DMODEL + h * DHEAD + d] = f2bf(o);
    }
}

// ---------------------------------------------------------------------------
extern "C" void kernel_launch(void* const* d_in, const int* in_sizes, int n_in,
                              void* d_out, int out_size, void* d_ws, size_t ws_size,
                              hipStream_t stream) {
  const float* x  = (const float*)d_in[0];
  const float* qw = (const float*)d_in[1];
  const float* kw = (const float*)d_in[2];
  const float* vw = (const float*)d_in[3];
  const float* ow = (const float*)d_in[4];
  float* out = (float*)d_out;

  // Workspace: 4 bf16 buffers of 8192*1024 = 16 MB each (64 MB total).
  uint16_t* Q  = (uint16_t*)d_ws;                 // [b,h,s,dh]
  uint16_t* Kb = Q  + (size_t)MROWS * DMODEL;     // [b,h,s,dh]
  uint16_t* Vt = Kb + (size_t)MROWS * DMODEL;     // [b,h,dh,s]
  uint16_t* Ob = Vt + (size_t)MROWS * DMODEL;     // [b,s,h*dh]

  dim3 ggrid(DMODEL / 64, MROWS / 64);
  gemm_bf16_wmma<<<ggrid, 128, 0, stream>>>(x, 0, qw, Q,  0);
  gemm_bf16_wmma<<<ggrid, 128, 0, stream>>>(x, 0, kw, Kb, 0);
  gemm_bf16_wmma<<<ggrid, 128, 0, stream>>>(x, 0, vw, Vt, 1);

  const int rope_threads = 64 * SEQ * 32;         // one thread per rotation pair
  rope_kernel<<<rope_threads / 256, 256, 0, stream>>>(Q);
  rope_kernel<<<rope_threads / 256, 256, 0, stream>>>(Kb);

  attn_kernel<<<64 * (SEQ / 16), 32, 0, stream>>>(Q, Kb, Vt, Ob);

  gemm_bf16_wmma<<<ggrid, 128, 0, stream>>>(Ob, 1, ow, out, 2);
}
